// MultiHeadAttention_40011915329967
// MI455X (gfx1250) — matrix operand order repaired and checked by reading, not compiled, not run on hardware
//
#include <hip/hip_runtime.h>
#include <hip/hip_bf16.h>

typedef __attribute__((ext_vector_type(4)))  _Float16 v4h;
typedef __attribute__((ext_vector_type(8)))  _Float16 v8h;
typedef __attribute__((ext_vector_type(16))) _Float16 v16h;
typedef __attribute__((ext_vector_type(8)))  float    v8f;

constexpr int NB = 4;       // batch
constexpr int TT = 2048;    // seq len
constexpr int DM = 1024;    // d_model
constexpr int NH = 16;      // heads
constexpr int DH = 64;      // head dim
constexpr int ROWS = NB * TT; // 8192

__device__ __forceinline__ v16h cat8(v8h lo, v8h hi) {
  return __builtin_shufflevector(lo, hi, 0,1,2,3,4,5,6,7,8,9,10,11,12,13,14,15);
}

// A-operand fragment (16xK tile, K-chunk of 32), row-major LDS tile, stride in halves.
// lane<16: K 0..15; lane>=16: K 16..31 (matches frag_b)
__device__ __forceinline__ v16h frag_a(const _Float16* tile, int row, int strideH, int lh) {
  const _Float16* p = tile + row * strideH + lh * 16;
  return cat8(*(const v8h*)p, *(const v8h*)(p + 8));
}
// B-operand fragment: LDS tile stored N-major (tile[col][k]); lane<16: K 0..15,
// lane>=16: K 16..31 -> one contiguous 32B run per lane.
__device__ __forceinline__ v16h frag_b(const _Float16* tile, int col, int strideH, int lh) {
  const _Float16* p = tile + col * strideH + lh * 16;
  return cat8(*(const v8h*)p, *(const v8h*)(p + 8));
}
__device__ __forceinline__ v8f wmma16(v16h a, v16h b, v8f c) {
  return __builtin_amdgcn_wmma_f32_16x16x32_f16(false, a, false, b, (short)0, c, false, false);
}

// CDNA5 async copy: 16B per active lane, global -> LDS, tracked by ASYNCcnt.
// LDS byte address = low 32 bits of the generic pointer (flat LDS aperture
// maps addr[31:0] directly onto the wave's LDS allocation).
__device__ __forceinline__ void async_ld16(void* lds, const void* g) {
  unsigned loff = (unsigned)(uintptr_t)lds;
  asm volatile("global_load_async_to_lds_b128 %0, %1, off"
               :: "v"(loff), "v"(g) : "memory");
}
__device__ __forceinline__ void wait_async0() {
  asm volatile("s_wait_asynccnt 0" ::: "memory");
}

// ---------------------------------------------------------------------------
// Tiled GEMM: [ROWS x DM] @ [DM x DM] + bias.  64x64 tile per 128-thread block.
// OUT_MODE 0: f16 out as [N,H,T,D]; 1: f16 out transposed [N,H,D,T]; 2: f32 row-major -> d_out
// A_F16: A operand already f16 (context; staged via async-to-LDS) vs f32 (x inputs).
// ---------------------------------------------------------------------------
template<int OUT_MODE, bool A_F16>
__global__ void __launch_bounds__(128)
gemm_wmma(const void* __restrict__ Ain, const float* __restrict__ B,
          const float* __restrict__ bias, void* __restrict__ Out, float out_scale)
{
  __shared__ _Float16 As[64 * 32];   // row-major [row][k]
  __shared__ _Float16 Bs[64 * 32];   // N-major  [col][k]  (= B^T tile)
  const int t    = threadIdx.x;
  const int lane = t & 31, wave = t >> 5;
  const int ln   = lane & 15, lh = lane >> 4;
  const int wr   = (wave >> 1) * 32, wc = (wave & 1) * 32;
  const int rb   = blockIdx.y * 64,  cb = blockIdx.x * 64;
  const float*    Af = (const float*)Ain;
  const _Float16* Ah = (const _Float16*)Ain;

  v8f acc[2][2] = {};
  for (int kb = 0; kb < DM; kb += 32) {
    // stage A tile 64x32
    if (A_F16) {
      // direct f16 -> LDS async copy: 256 x 16B chunks, 2 per thread
      #pragma unroll
      for (int i = 0; i < 2; ++i) {
        int c   = t * 2 + i;            // chunk id 0..255
        int row = c >> 2;
        int kq  = (c & 3) * 8;
        async_ld16(&As[row * 32 + kq], Ah + (size_t)(rb + row) * DM + kb + kq);
      }
    } else {
      #pragma unroll
      for (int rr = 0; rr < 4; ++rr) {
        int row = rr * 16 + (t >> 3);
        int kq  = (t & 7) * 4;
        size_t g = (size_t)(rb + row) * DM + kb + kq;
        float4 a4 = *(const float4*)(Af + g);
        As[row*32+kq+0] = (_Float16)a4.x;
        As[row*32+kq+1] = (_Float16)a4.y;
        As[row*32+kq+2] = (_Float16)a4.z;
        As[row*32+kq+3] = (_Float16)a4.w;
        if (kb + 32 < DM) __builtin_prefetch(Af + g + 32, 0, 1);
      }
    }
    // stage B^T tile 64 cols x 32 k (f32 -> f16 convert, transposed scatter)
    #pragma unroll
    for (int rr = 0; rr < 4; ++rr) {
      int k = rr * 8 + (t >> 4);
      int c = (t & 15) * 4;
      size_t g = (size_t)(kb + k) * DM + cb + c;
      float4 b4 = *(const float4*)(B + g);
      Bs[(c+0)*32 + k] = (_Float16)b4.x;
      Bs[(c+1)*32 + k] = (_Float16)b4.y;
      Bs[(c+2)*32 + k] = (_Float16)b4.z;
      Bs[(c+3)*32 + k] = (_Float16)b4.w;
      if (kb + 32 < DM) __builtin_prefetch(B + g + (size_t)32 * DM, 0, 1);
    }
    if (A_F16) wait_async0();
    __syncthreads();
    v16h a0 = frag_a(As, wr + ln,      32, lh);
    v16h a1 = frag_a(As, wr + 16 + ln, 32, lh);
    v16h b0 = frag_b(Bs, wc + ln,      32, lh);
    v16h b1 = frag_b(Bs, wc + 16 + ln, 32, lh);
    acc[0][0] = wmma16(a0, b0, acc[0][0]);
    acc[0][1] = wmma16(a0, b1, acc[0][1]);
    acc[1][0] = wmma16(a1, b0, acc[1][0]);
    acc[1][1] = wmma16(a1, b1, acc[1][1]);
    __syncthreads();
  }
  // epilogue: bias (+ optional scale), scatter per output layout
  #pragma unroll
  for (int j = 0; j < 2; ++j) {
    int gcol = cb + wc + j * 16 + ln;
    float bj = bias[gcol];
    #pragma unroll
    for (int i = 0; i < 2; ++i) {
      #pragma unroll
      for (int r = 0; r < 8; ++r) {
        int grow = rb + wr + i * 16 + r + lh * 8;
        float v = (acc[i][j][r] + bj) * out_scale;
        if (OUT_MODE == 2) {
          ((float*)Out)[(size_t)grow * DM + gcol] = v;
        } else {
          int n = grow >> 11, tt = grow & (TT - 1);
          int h = gcol >> 6,  d  = gcol & 63;
          size_t off;
          if (OUT_MODE == 0) off = ((size_t)(n * NH + h) * TT + tt) * DH + d;
          else               off = ((size_t)(n * NH + h) * DH + d) * TT + tt;
          ((_Float16*)Out)[off] = (_Float16)v;
        }
      }
    }
  }
}

// ---------------------------------------------------------------------------
// Causal flash attention. One block = 64 q rows of one (n,h); 4 waves x 16 rows.
// Q/K/V tiles staged with GLOBAL_LOAD_ASYNC_TO_LDS (ASYNCcnt).
// Q pre-scaled by 1/sqrt(D) in the projection.
// ---------------------------------------------------------------------------
__global__ void __launch_bounds__(128)
flash_attn(const _Float16* __restrict__ Qh, const _Float16* __restrict__ Kh,
           const _Float16* __restrict__ Vt, const int* __restrict__ pmask,
           _Float16* __restrict__ Ctx)
{
  __shared__ _Float16 Qs[64 * 64];     // [qrow][d]
  __shared__ _Float16 Ks[32 * 64];     // [key][d]   (== B^T for scores)
  __shared__ _Float16 Vs[64 * 32];     // [d][key]   (== B^T for PV)
  __shared__ _Float16 Ps[4][16 * 32];  // per-wave P scratch
  __shared__ float    pmb[32];         // padding bias per key

  const int t    = threadIdx.x;
  const int lane = t & 31, wave = t >> 5;
  const int ln   = lane & 15, lh = lane >> 4;
  const int qt0  = blockIdx.x * 64;
  const int bh   = blockIdx.y;             // n*NH + h
  const int n    = bh >> 4, h = bh & 15;
  const size_t headBase = (size_t)bh * TT * DH;
  const int qb   = qt0 + wave * 16;        // this wave's first absolute q row

  { // stage Q once (contiguous 8 KB) via async copy
    const _Float16* qg = Qh + headBase + (size_t)qt0 * DH;
    #pragma unroll
    for (int i = 0; i < 4; ++i) {
      int off = (t * 4 + i) * 8;
      async_ld16(Qs + off, qg + off);
    }
  }
  v8f acc[4] = {};
  float m[8], l[8];
  #pragma unroll
  for (int r = 0; r < 8; ++r) { m[r] = -__builtin_inff(); l[r] = 0.f; }

  const int nkt = qt0 / 32 + 2;            // causal clip at tile granularity
  for (int kt = 0; kt < nkt; ++kt) {
    { // stage K tile (contiguous) and V^T tile (row-strided) via async copy
      const _Float16* kg = Kh + headBase + (size_t)(kt * 32) * DH;
      #pragma unroll
      for (int i = 0; i < 2; ++i) {
        int off = (t * 2 + i) * 8;
        async_ld16(Ks + off, kg + off);
      }
      int d = t >> 1, part = (t & 1) * 16;
      const _Float16* vg = Vt + (size_t)bh * DH * TT + (size_t)d * TT + kt * 32 + part;
      async_ld16(Vs + d * 32 + part,     vg);
      async_ld16(Vs + d * 32 + part + 8, vg + 8);
      if (t < 32) pmb[t] = pmask[n * TT + kt * 32 + t] ? 0.f : -__builtin_inff();
      if (kt + 1 < nkt) __builtin_prefetch(kg + 32 * DH, 0, 1);
    }
    wait_async0();                         // Q (first iter) + K/V in LDS
    __syncthreads();

    if (kt * 32 <= qb + 15) {              // wave needs this key tile (causal)
      v8f s0 = {}, s1 = {};
      #pragma unroll
      for (int kc = 0; kc < 2; ++kc) {     // D = 64 -> two K-chunks of 32
        v16h aq = frag_a(Qs + kc * 32, wave * 16 + ln, 64, lh);
        v16h b0 = frag_b(Ks + kc * 32, ln,      64, lh);
        v16h b1 = frag_b(Ks + kc * 32, 16 + ln, 64, lh);
        s0 = wmma16(aq, b0, s0);
        s1 = wmma16(aq, b1, s1);
      }
      const int k0 = kt * 32 + ln, k1 = k0 + 16;
      const float pb0 = pmb[ln], pb1 = pmb[16 + ln];
      _Float16* Pw = &Ps[wave][0];
      #pragma unroll
      for (int r = 0; r < 8; ++r) {
        int qrow = qb + r + lh * 8;
        float v0 = s0[r] + pb0; if (k0 > qrow) v0 = -__builtin_inff();
        float v1 = s1[r] + pb1; if (k1 > qrow) v1 = -__builtin_inff();
        // row max across the 16-lane half (rows differ between halves)
        float rm = fmaxf(v0, v1);
        rm = fmaxf(rm, __shfl_xor(rm, 1, 32));
        rm = fmaxf(rm, __shfl_xor(rm, 2, 32));
        rm = fmaxf(rm, __shfl_xor(rm, 4, 32));
        rm = fmaxf(rm, __shfl_xor(rm, 8, 32));
        float mn  = fmaxf(m[r], rm);
        bool  dead = (mn == -__builtin_inff());
        float alpha = dead ? 1.f : __expf(m[r] - mn);
        float p0 = dead ? 0.f : __expf(v0 - mn);
        float p1 = dead ? 0.f : __expf(v1 - mn);
        float rs = p0 + p1;
        rs += __shfl_xor(rs, 1, 32);
        rs += __shfl_xor(rs, 2, 32);
        rs += __shfl_xor(rs, 4, 32);
        rs += __shfl_xor(rs, 8, 32);
        m[r] = mn;
        l[r] = l[r] * alpha + rs;
        #pragma unroll
        for (int dt = 0; dt < 4; ++dt) acc[dt][r] *= alpha;
        Pw[(r + lh * 8) * 32 + ln]      = (_Float16)p0;  // C-layout -> LDS
        Pw[(r + lh * 8) * 32 + 16 + ln] = (_Float16)p1;
      }
      // reload P in A-operand layout (same wave, DS kept in order by compiler)
      v16h ap = frag_a(Pw, ln, 32, lh);
      #pragma unroll
      for (int dt = 0; dt < 4; ++dt) {
        v16h bv = frag_b(Vs, dt * 16 + ln, 32, lh);
        acc[dt] = wmma16(ap, bv, acc[dt]);
      }
    }
    __syncthreads();
  }

  // normalize and write context [N,T,H*D] (f16, A-operand for out-proj GEMM)
  #pragma unroll
  for (int dt = 0; dt < 4; ++dt) {
    #pragma unroll
    for (int r = 0; r < 8; ++r) {
      int qrow = qb + r + lh * 8;
      float ov = acc[dt][r] / l[r];
      Ctx[(size_t)(n * TT + qrow) * (NH * DH) + h * DH + dt * 16 + ln] = (_Float16)ov;
    }
  }
}

extern "C" void kernel_launch(void* const* d_in, const int* in_sizes, int n_in,
                              void* d_out, int out_size, void* d_ws, size_t ws_size,
                              hipStream_t stream) {
  const float* xq = (const float*)d_in[0];
  const float* xk = (const float*)d_in[1];
  const float* xv = (const float*)d_in[2];
  const int*   pm = (const int*)  d_in[3];
  const float* Wq = (const float*)d_in[4];
  const float* bq = (const float*)d_in[5];
  const float* Wk = (const float*)d_in[6];
  const float* bk = (const float*)d_in[7];
  const float* Wv = (const float*)d_in[8];
  const float* bv = (const float*)d_in[9];
  const float* Wo = (const float*)d_in[10];
  const float* bo = (const float*)d_in[11];

  const size_t HE = (size_t)NB * NH * TT * DH;   // 8M elems per tensor
  _Float16* Qh  = (_Float16*)d_ws;               // [N,H,T,D]
  _Float16* Kh  = Qh + HE;                       // [N,H,T,D]
  _Float16* Vt  = Kh + HE;                       // [N,H,D,T]
  _Float16* Ctx = Vt + HE;                       // [N,T,H*D]

  dim3 gg(DM / 64, ROWS / 64);
  // Q projection carries the 1/sqrt(D) softmax scale (applied after bias).
  gemm_wmma<0, false><<<gg, 128, 0, stream>>>(xq, Wq, bq, Qh, 0.125f);
  gemm_wmma<0, false><<<gg, 128, 0, stream>>>(xk, Wk, bk, Kh, 1.0f);
  gemm_wmma<1, false><<<gg, 128, 0, stream>>>(xv, Wv, bv, Vt, 1.0f);
  flash_attn<<<dim3(TT / 64, NB * NH), 128, 0, stream>>>(Qh, Kh, Vt, pm, Ctx);
  gemm_wmma<2, true><<<gg, 128, 0, stream>>>(Ctx, Wo, bo, d_out, 1.0f);
}
